// Encode_36979668418631
// MI455X (gfx1250) — matrix operand order repaired and checked by reading, not compiled, not run on hardware
//
#include <hip/hip_runtime.h>
#include <math.h>
#include <stdint.h>

typedef __attribute__((ext_vector_type(16))) _Float16 v16h;
typedef __attribute__((ext_vector_type(8)))  float    v8f;
typedef long long ll;

#define EPI_ACCUM 1
#define EPI_BIAS  2
#define EPI_RESID 4
#define EPI_RELU  8

// LDS byte-offset of a __shared__ object (generic -> addrspace(3) cast).
#define LDS_OFF(p) ((uint32_t)(uintptr_t)(const __attribute__((address_space(3))) void*)(p))

// ---------------------------------------------------------------------------
// WMMA fragment helpers (layouts per CDNA5 ISA 7.12.2, wave32)
// A 16x32 f16: lane m=lane&15, g=lane>>4; halves j<8 -> K=j+8g ; j>=8 -> K=j+8+8g
// B 32x16 f16: lane n=lane&15, g=lane>>4; same K order as A (lds_b_frag); flash-attn A/B both use K=j+16g
// C/D 16x16 f32: VGPR r -> row r+8g, col = lane&15
// ---------------------------------------------------------------------------
__device__ __forceinline__ v16h pack16(float4 x0, float4 x1, float4 x2, float4 x3,
                                       float scale) {
  v16h a;
  a[0]=(_Float16)(x0.x*scale); a[1]=(_Float16)(x0.y*scale);
  a[2]=(_Float16)(x0.z*scale); a[3]=(_Float16)(x0.w*scale);
  a[4]=(_Float16)(x1.x*scale); a[5]=(_Float16)(x1.y*scale);
  a[6]=(_Float16)(x1.z*scale); a[7]=(_Float16)(x1.w*scale);
  a[8]=(_Float16)(x2.x*scale); a[9]=(_Float16)(x2.y*scale);
  a[10]=(_Float16)(x2.z*scale); a[11]=(_Float16)(x2.w*scale);
  a[12]=(_Float16)(x3.x*scale); a[13]=(_Float16)(x3.y*scale);
  a[14]=(_Float16)(x3.z*scale); a[15]=(_Float16)(x3.w*scale);
  return a;
}

__device__ __forceinline__ v16h load_a_rowptr(const float* __restrict__ row,
                                              int k0, int g, float scale) {
  const float* p = row + k0 + 16 * g;
  return pack16(((const float4*)p)[0], ((const float4*)p)[1],
                ((const float4*)(p + 8))[0], ((const float4*)(p + 8))[1], scale);
}

__device__ __forceinline__ v16h load16_contig(const float* __restrict__ p) {
  return pack16(((const float4*)p)[0], ((const float4*)p)[1],
                ((const float4*)p)[2], ((const float4*)p)[3], 1.0f);
}

// A fragment out of an LDS tile row (row stride fixed by caller).
__device__ __forceinline__ v16h lds_a_frag(const float* ar, int g) {
  float4 x0 = *(const float4*)(ar + 8 * g);
  float4 x1 = *(const float4*)(ar + 8 * g + 4);
  float4 x2 = *(const float4*)(ar + 16 + 8 * g);
  float4 x3 = *(const float4*)(ar + 20 + 8 * g);
  return pack16(x0, x1, x2, x3, 1.0f);
}

#define BS_STRIDE 132  // 128 + 4 pad: break bank aliasing on 64-bank LDS
#define AS_STRIDE 36   // 32 + 4 pad

__device__ __forceinline__ v16h lds_b_frag(const float* Bs, int col, int g) {
  const float* p = Bs + col;
  v16h b;
#pragma unroll
  for (int j = 0; j < 16; ++j) b[j] = (_Float16)p[(size_t)(j < 8 ? j + 8 * g : j + 8 + 8 * g) * BS_STRIDE];
  return b;
}

// ---------------------------------------------------------------------------
// Generic batched GEMM:  C[b] = A[b](MxK) * B[b](KxN)  (+epilogue)
// 256 threads = 8 waves; wave tile 16x64 (4 wmma/N), block tile 64x128x32.
// A/B tiles staged in LDS via GLOBAL_LOAD_ASYNC_TO_LDS (ASYNCcnt DMA path),
// s_wait_asynccnt + workgroup barrier, then WMMA fragments fed from LDS.
// All shapes in this model are exact multiples of the 64x128 block tile.
// BVEC=1: B rows 16B-aligned -> async b128; BVEC=0 (k3 taps): async b32.
// ---------------------------------------------------------------------------
template <int EPI, int BVEC>
__global__ __launch_bounds__(256) void gemm_wmma_kernel(
    const float* __restrict__ A, int lda, ll sA,
    const float* __restrict__ B, int ldb, ll sB,
    float* __restrict__ C, int ldc, ll sC,
    const float* __restrict__ bias,
    const float* __restrict__ resid, ll sR,
    int M, int N, int K) {
  __shared__ float As[64 * AS_STRIDE];
  __shared__ float Bs[32 * BS_STRIDE];
  const int tid = threadIdx.x;
  const int lane = tid & 31;
  const int wave = tid >> 5;
  const int rowBase = blockIdx.x * 64;     // block M origin
  const int colBase = blockIdx.y * 128;    // block N origin
  const int tileMl = (wave & 3) * 16;      // wave M origin within block
  const int tileNl = (wave >> 2) * 64;     // wave N origin within block
  const int bz = blockIdx.z;
  const float* Ab = A + (size_t)bz * sA;
  const float* Bb = B + (size_t)bz * sB;
  float* Cb = C + (size_t)bz * sC;
  const int n = lane & 15, g = lane >> 4;

  v8f acc[4];
#pragma unroll
  for (int t = 0; t < 4; ++t) {
    if (EPI & EPI_ACCUM) {
#pragma unroll
      for (int r = 0; r < 8; ++r)
        acc[t][r] = Cb[(size_t)(rowBase + tileMl + r + 8 * g) * ldc +
                       colBase + tileNl + t * 16 + n];
    } else {
#pragma unroll
      for (int r = 0; r < 8; ++r) acc[t][r] = 0.0f;
    }
  }

  const float* arow = &As[(size_t)(tileMl + (lane & 15)) * AS_STRIDE];

  for (int k0 = 0; k0 < K; k0 += 32) {
    // ---- cooperative async DMA of A (64x32) and B (32x128) into LDS ----
#pragma unroll
    for (int i = 0; i < 2; ++i) {           // A: 512 float4
      int idx = tid + 256 * i;
      int row = idx >> 3, c4 = (idx & 7) * 4;
      uint64_t ga = (uint64_t)(uintptr_t)(Ab + (size_t)(rowBase + row) * lda + k0 + c4);
      uint32_t lo = LDS_OFF(&As[(size_t)row * AS_STRIDE + c4]);
      asm volatile("global_load_async_to_lds_b128 %0, %1, off"
                   :: "v"(lo), "v"(ga) : "memory");
    }
    if (BVEC) {
#pragma unroll
      for (int i = 0; i < 4; ++i) {         // B: 1024 float4
        int idx = tid + 256 * i;
        int row = idx >> 5, c4 = (idx & 31) * 4;
        uint64_t ga = (uint64_t)(uintptr_t)(Bb + (size_t)(k0 + row) * ldb + colBase + c4);
        uint32_t lo = LDS_OFF(&Bs[(size_t)row * BS_STRIDE + c4]);
        asm volatile("global_load_async_to_lds_b128 %0, %1, off"
                     :: "v"(lo), "v"(ga) : "memory");
      }
    } else {
#pragma unroll
      for (int i = 0; i < 16; ++i) {        // B: 4096 b32 (unaligned-safe)
        int idx = tid + 256 * i;
        int row = idx >> 7, c = idx & 127;
        uint64_t ga = (uint64_t)(uintptr_t)(Bb + (size_t)(k0 + row) * ldb + colBase + c);
        uint32_t lo = LDS_OFF(&Bs[(size_t)row * BS_STRIDE + c]);
        asm volatile("global_load_async_to_lds_b32 %0, %1, off"
                     :: "v"(lo), "v"(ga) : "memory");
      }
    }
    asm volatile("s_wait_asynccnt 0x0" ::: "memory");
    __syncthreads();

    // ---- WMMA from LDS tiles ----
    v16h a = lds_a_frag(arow, g);
#pragma unroll
    for (int t = 0; t < 4; ++t) {
      v16h bf = lds_b_frag(Bs, tileNl + t * 16 + n, g);
      acc[t] = __builtin_amdgcn_wmma_f32_16x16x32_f16(false, a, false, bf,
                                                      (short)0, acc[t], false, false);
    }
    __syncthreads();
  }

#pragma unroll
  for (int t = 0; t < 4; ++t) {
#pragma unroll
    for (int r = 0; r < 8; ++r) {
      size_t row = (size_t)(rowBase + tileMl + r + 8 * g);
      size_t col = (size_t)(colBase + tileNl + t * 16 + n);
      float v = acc[t][r];
      if (EPI & EPI_BIAS)  v += bias[col];
      if (EPI & EPI_RESID) v += resid[(size_t)bz * sR + row * ldc + col];
      if (EPI & EPI_RELU)  v = fmaxf(v, 0.0f);
      Cb[row * ldc + col] = v;
    }
  }
}

static void gemm(hipStream_t st, const float* A, int lda, ll sA,
                 const float* B, int ldb, ll sB,
                 float* C, int ldc, ll sC,
                 int M, int N, int K, int batch, int epi, int bvec,
                 const float* bias = nullptr, const float* resid = nullptr,
                 ll sR = 0) {
  dim3 grid((M + 63) / 64, (N + 127) / 128, batch), blk(256);
#define GLAUNCH(E, BV) \
  gemm_wmma_kernel<E, BV><<<grid, blk, 0, st>>>(A, lda, sA, B, ldb, sB, C, ldc, sC, \
                                                bias, resid, sR, M, N, K)
  if (bvec) {
    switch (epi) {
      case 0:  GLAUNCH(0, 1); break;
      case 4:  GLAUNCH(4, 1); break;
      case 6:  GLAUNCH(6, 1); break;
      case 10: GLAUNCH(10, 1); break;
      case 12: GLAUNCH(12, 1); break;
      default: GLAUNCH(0, 1); break;
    }
  } else {
    switch (epi) {
      case 1:  GLAUNCH(1, 0); break;
      default: GLAUNCH(0, 0); break;
    }
  }
#undef GLAUNCH
}

// ---------------------------------------------------------------------------
// Flash attention: one wave per (b, head, 16-query tile). hd=64, no mask.
// Q pre-scaled by 1/sqrt(64)=0.125 at f16 conversion. Scores via 2 WMMA per
// 16-key subtile; P re-laid out through LDS (DS in-order per wave +
// s_wait_dscnt) into an A-fragment; P*V via 4 WMMA per 32-key chunk.
// ---------------------------------------------------------------------------
__global__ __launch_bounds__(32) void flash_attn_kernel(
    const float* __restrict__ Q, const float* __restrict__ K,
    const float* __restrict__ V, float* __restrict__ O, int S) {
  __shared__ _Float16 psh[16 * 32];
  const int lane = threadIdx.x;
  const int qt = blockIdx.x, h = blockIdx.y, b = blockIdx.z;
  const int n = lane & 15, g = lane >> 4;
  const float* Qb = Q + ((size_t)b * S) * 1024 + (size_t)h * 64;
  const float* Kb = K + ((size_t)b * S) * 1024 + (size_t)h * 64;
  const float* Vb = V + ((size_t)b * S) * 1024 + (size_t)h * 64;

  const float* qrow = Qb + (size_t)(qt * 16 + n) * 1024;  // m == lane&15
  v16h aq0 = load_a_rowptr(qrow, 0, g, 0.125f);
  v16h aq1 = load_a_rowptr(qrow, 32, g, 0.125f);

  float mrow[8], lrow[8];
  v8f o[4];
#pragma unroll
  for (int r = 0; r < 8; ++r) {
    mrow[r] = -3.0e38f; lrow[r] = 0.0f;
    o[0][r] = 0.f; o[1][r] = 0.f; o[2][r] = 0.f; o[3][r] = 0.f;
  }

  for (int kc = 0; kc < S; kc += 32) {
    v8f s0, s1;
    {
      const float* kr = Kb + (size_t)(kc + n) * 1024;  // keys kc..kc+15
      v8f z;
#pragma unroll
      for (int r = 0; r < 8; ++r) z[r] = 0.f;
      v16h bk0 = load16_contig(kr + 16 * g);
      v16h bk1 = load16_contig(kr + 32 + 16 * g);
      s0 = __builtin_amdgcn_wmma_f32_16x16x32_f16(false, aq0, false, bk0, (short)0, z, false, false);
      s0 = __builtin_amdgcn_wmma_f32_16x16x32_f16(false, aq1, false, bk1, (short)0, s0, false, false);
    }
    {
      const float* kr = Kb + (size_t)(kc + 16 + n) * 1024;  // keys kc+16..kc+31
      v8f z;
#pragma unroll
      for (int r = 0; r < 8; ++r) z[r] = 0.f;
      v16h bk0 = load16_contig(kr + 16 * g);
      v16h bk1 = load16_contig(kr + 32 + 16 * g);
      s1 = __builtin_amdgcn_wmma_f32_16x16x32_f16(false, aq0, false, bk0, (short)0, z, false, false);
      s1 = __builtin_amdgcn_wmma_f32_16x16x32_f16(false, aq1, false, bk1, (short)0, s1, false, false);
    }

    // online softmax per row (row r+8g lives across the 16 lanes of group g)
#pragma unroll
    for (int r = 0; r < 8; ++r) {
      float v = fmaxf(s0[r], s1[r]);
#pragma unroll
      for (int off = 8; off >= 1; off >>= 1) v = fmaxf(v, __shfl_xor(v, off, 32));
      float nm = fmaxf(mrow[r], v);
      float alpha = __expf(mrow[r] - nm);
      float p0 = __expf(s0[r] - nm);
      float p1 = __expf(s1[r] - nm);
      float rs = p0 + p1;
#pragma unroll
      for (int off = 8; off >= 1; off >>= 1) rs += __shfl_xor(rs, off, 32);
      lrow[r] = lrow[r] * alpha + rs;
      mrow[r] = nm;
      o[0][r] *= alpha; o[1][r] *= alpha; o[2][r] *= alpha; o[3][r] *= alpha;
      psh[(r + 8 * g) * 32 + n] = (_Float16)p0;
      psh[(r + 8 * g) * 32 + n + 16] = (_Float16)p1;
    }
    // DS ops are in-order per wave on CDNA5; wait out the split DS counter.
    asm volatile("s_wait_dscnt 0x0" ::: "memory");

    v16h ap;  // P as 16x32 A-fragment
#pragma unroll
    for (int j = 0; j < 8; ++j)  ap[j] = psh[n * 32 + j + 16 * g];
#pragma unroll
    for (int j = 8; j < 16; ++j) ap[j] = psh[n * 32 + j + 16 * g];

#pragma unroll
    for (int t = 0; t < 4; ++t) {
      v16h bv;
      const float* vb = Vb + t * 16 + n;
#pragma unroll
      for (int j = 0; j < 16; ++j)
        bv[j] = (_Float16)vb[(size_t)(kc + j + 16 * g) * 1024];
      o[t] = __builtin_amdgcn_wmma_f32_16x16x32_f16(false, ap, false, bv, (short)0, o[t], false, false);
    }
  }

#pragma unroll
  for (int t = 0; t < 4; ++t)
#pragma unroll
    for (int r = 0; r < 8; ++r)
      O[((size_t)b * S + qt * 16 + r + 8 * g) * 1024 + h * 64 + t * 16 + n] =
          o[t][r] / lrow[r];
}

// ---------------------------------------------------------------------------
// Support kernels
// ---------------------------------------------------------------------------
__global__ void stem_conv_kernel(const float* __restrict__ x,
                                 const float* __restrict__ w0, const float* __restrict__ w1,
                                 const float* __restrict__ w2, const float* __restrict__ w3,
                                 float* __restrict__ out, int L) {
  int i = blockIdx.x * blockDim.x + threadIdx.x;
  int total = 4 * 64 * L;
  if (i >= total) return;
  int l = i % L, co = (i / L) & 63, b = i / (L * 64);
  int grp = co >> 4, colc = co & 15;
  int ksz = 2 * grp + 1, pad = grp;
  const float* w = (grp == 0) ? w0 : (grp == 1) ? w1 : (grp == 2) ? w2 : w3;
  const float4* x4 = (const float4*)x;
  float acc = 0.f;
  for (int t = 0; t < ksz; ++t) {
    int ll = l + t - pad;
    if (ll < 0 || ll >= L) continue;
    float4 xv = x4[(size_t)b * L + ll];
    const float* wt = w + (size_t)(colc * 4) * ksz + t;
    acc += xv.x * wt[0] + xv.y * wt[ksz] + xv.z * wt[2 * ksz] + xv.w * wt[3 * ksz];
  }
  out[((size_t)b * 64 + co) * L + l] = acc;
}

__global__ __launch_bounds__(128) void ln_ch_kernel(float* __restrict__ x,
                                                    const float* __restrict__ g,
                                                    const float* __restrict__ b,
                                                    int C, int L) {
  __shared__ float s1[128], s2[128];
  int bl = blockIdx.x;
  int bb = bl / L, l = bl - bb * L;
  float* base = x + (size_t)bb * C * L + l;
  int tid = threadIdx.x;
  float sum = 0.f, sq = 0.f;
  for (int c = tid; c < C; c += 128) {
    float v = base[(size_t)c * L];
    sum += v; sq += v * v;
  }
  s1[tid] = sum; s2[tid] = sq; __syncthreads();
  for (int off = 64; off; off >>= 1) {
    if (tid < off) { s1[tid] += s1[tid + off]; s2[tid] += s2[tid + off]; }
    __syncthreads();
  }
  float mean = s1[0] / C;
  float var = s2[0] / C - mean * mean;
  float rstd = rsqrtf(var + 1e-5f);
  for (int c = tid; c < C; c += 128) {
    float v = base[(size_t)c * L];
    base[(size_t)c * L] = (v - mean) * rstd * g[c] + b[c];
  }
}

__global__ __launch_bounds__(256) void row_ln_kernel(const float* __restrict__ in,
                                                     float* __restrict__ out,
                                                     const float* __restrict__ g,
                                                     const float* __restrict__ b,
                                                     int D, int mode) {
  __shared__ float s1[256], s2[256];
  int row = blockIdx.x, tid = threadIdx.x;
  const float* p = in + (size_t)row * D;
  float* q = out + (size_t)row * D;
  float sum = 0.f, sq = 0.f;
  for (int i = tid; i < D; i += 256) { float v = p[i]; sum += v; sq += v * v; }
  s1[tid] = sum; s2[tid] = sq; __syncthreads();
  for (int off = 128; off; off >>= 1) {
    if (tid < off) { s1[tid] += s1[tid + off]; s2[tid] += s2[tid + off]; }
    __syncthreads();
  }
  float mean = s1[0] / D;
  if (mode == 0) {
    float var = s2[0] / D - mean * mean;
    float rstd = rsqrtf(var + 1e-5f);
    for (int i = tid; i < D; i += 256) q[i] = (p[i] - mean) * rstd * g[i] + b[i];
  } else {
    float var = (s2[0] - D * mean * mean) / (D - 1);
    float inv = 1.0f / (sqrtf(var) + 1e-6f);
    for (int i = tid; i < D; i += 256) q[i] = (p[i] - mean) * inv;
  }
}

__global__ __launch_bounds__(256) void bn_stats_kernel(const float* __restrict__ x,
                                                       const float* __restrict__ g,
                                                       const float* __restrict__ b,
                                                       float* __restrict__ scale,
                                                       float* __restrict__ shift,
                                                       int C, int L) {
  __shared__ float s1[256], s2[256];
  int c = blockIdx.x, tid = threadIdx.x;
  float sum = 0.f, sq = 0.f;
  int N = 4 * L;
  for (int i = tid; i < N; i += 256) {
    int bb = i / L, l = i - bb * L;
    float v = x[((size_t)bb * C + c) * L + l];
    sum += v; sq += v * v;
  }
  s1[tid] = sum; s2[tid] = sq; __syncthreads();
  for (int off = 128; off; off >>= 1) {
    if (tid < off) { s1[tid] += s1[tid + off]; s2[tid] += s2[tid + off]; }
    __syncthreads();
  }
  if (tid == 0) {
    float mean = s1[0] / N;
    float var = s2[0] / N - mean * mean;
    float sc = g[c] * rsqrtf(var + 1e-5f);
    scale[c] = sc;
    shift[c] = b[c] - mean * sc;
  }
}

__global__ void bn_apply_relu_kernel(float* __restrict__ x, const float* __restrict__ scale,
                                     const float* __restrict__ shift, int C, int L) {
  int i = blockIdx.x * blockDim.x + threadIdx.x;
  int total = 4 * C * L;
  if (i >= total) return;
  int c = (i / L) % C;
  x[i] = fmaxf(x[i] * scale[c] + shift[c], 0.0f);
}

__global__ void relu_copy_kernel(const float* __restrict__ in, float* __restrict__ out, int n) {
  int i = blockIdx.x * blockDim.x + threadIdx.x;
  if (i < n) out[i] = fmaxf(in[i], 0.0f);
}

__global__ void add_relu_kernel(const float* __restrict__ a, float* __restrict__ io, int n) {
  int i = blockIdx.x * blockDim.x + threadIdx.x;
  if (i < n) io[i] = fmaxf(a[i] + io[i], 0.0f);
}

__global__ void pad_kernel(const float* __restrict__ x, float* __restrict__ out, int C, int L) {
  int i = blockIdx.x * blockDim.x + threadIdx.x;
  int total = 4 * C * (L + 2);
  if (i >= total) return;
  int j = i % (L + 2);
  int bc = i / (L + 2);
  out[i] = (j == 0 || j == L + 1) ? 0.0f : x[(size_t)bc * L + j - 1];
}

__global__ void repack_w2_kernel(const float* __restrict__ w, float* __restrict__ out, int ed) {
  int i = blockIdx.x * blockDim.x + threadIdx.x;
  int total = ed * ed * 3;
  if (i >= total) return;
  int t = i % 3;
  int ci = (i / 3) % ed;
  int co = i / (3 * ed);
  out[((size_t)t * ed + co) * ed + ci] = w[i];
}

__global__ void pool_combine_kernel(const float* __restrict__ x, const float* __restrict__ logits,
                                    float* __restrict__ out, int C, int L) {
  int i = blockIdx.x * blockDim.x + threadIdx.x;
  int half = L / 2;
  int total = 4 * C * half;
  if (i >= total) return;
  int n = i % half;
  size_t bc = (size_t)(i / half);
  float l0 = logits[bc * L + 2 * n], l1 = logits[bc * L + 2 * n + 1];
  float mx = fmaxf(l0, l1);
  float e0 = __expf(l0 - mx), e1 = __expf(l1 - mx);
  float x0 = x[bc * L + 2 * n], x1 = x[bc * L + 2 * n + 1];
  out[bc * half + n] = (x0 * e0 + x1 * e1) / (e0 + e1);
}

__global__ void crop_transpose_kernel(const float* __restrict__ in, float* __restrict__ out) {
  int i = blockIdx.x * blockDim.x + threadIdx.x;
  int total = 4 * 896 * 1024;
  if (i >= total) return;
  int c = i % 1024;
  int s = (i / 1024) % 896;
  int b = i / (1024 * 896);
  out[i] = in[((size_t)b * 1024 + c) * 1024 + 64 + s];
}

__global__ void rope_kernel(float* __restrict__ q, float* __restrict__ k, int S) {
  int i = blockIdx.x * blockDim.x + threadIdx.x;
  int total = 4 * S * 512;
  if (i >= total) return;
  int p = i & 511;
  int s = (i >> 9) % S;
  int b = i / (512 * S);
  int head = p >> 5, ii = p & 31;
  float inv = __expf(-(float)(2 * ii) * 0.14391156f);  // ln(1e4)/64
  float ang = (float)s * inv;
  float sn, cs;
  __sincosf(ang, &sn, &cs);
  size_t base = ((size_t)(b * S + s)) * 1024 + head * 64 + 2 * ii;
  float q0 = q[base], q1 = q[base + 1];
  q[base] = q0 * cs - q1 * sn;
  q[base + 1] = q1 * cs + q0 * sn;
  float k0 = k[base], k1 = k[base + 1];
  k[base] = k0 * cs - k1 * sn;
  k[base + 1] = k1 * cs + k0 * sn;
}

// ---------------------------------------------------------------------------
// Host orchestration
// ---------------------------------------------------------------------------
#define CEILDIV(a, b) (((a) + (b)-1) / (b))

static void run_bneck(hipStream_t st, const float* xin, int cin,
                      const float* w1, const float* g1, const float* b1,
                      const float* w2, const float* g2, const float* b2,
                      const float* w3, int ed, int L,
                      float* T1, float* T2, float* H3, float* PADB, float* WPK,
                      float* SCALE, float* SHIFT) {
  // t1 = w1 @ x  (M=ed, K=cin, N=L)
  gemm(st, w1, cin, 0, xin, L, (ll)cin * L, T1, L, (ll)ed * L, ed, L, cin, 4, 0, 1);
  bn_stats_kernel<<<ed, 256, 0, st>>>(T1, g1, b1, SCALE, SHIFT, ed, L);
  bn_apply_relu_kernel<<<CEILDIV(4 * ed * L, 256), 256, 0, st>>>(T1, SCALE, SHIFT, ed, L);
  // k3 conv: zero-padded input + 3 accumulating GEMMs with repacked weights
  pad_kernel<<<CEILDIV(4 * ed * (L + 2), 256), 256, 0, st>>>(T1, PADB, ed, L);
  repack_w2_kernel<<<CEILDIV(ed * ed * 3, 256), 256, 0, st>>>(w2, WPK, ed);
  for (int t = 0; t < 3; ++t)
    gemm(st, WPK + (size_t)t * ed * ed, ed, 0, PADB + t, L + 2, (ll)ed * (L + 2),
         T2, L, (ll)ed * L, ed, L, ed, 4, t ? EPI_ACCUM : 0, 0);
  bn_stats_kernel<<<ed, 256, 0, st>>>(T2, g2, b2, SCALE, SHIFT, ed, L);
  bn_apply_relu_kernel<<<CEILDIV(4 * ed * L, 256), 256, 0, st>>>(T2, SCALE, SHIFT, ed, L);
  // h3 = w3 @ t2  (M=4ed)
  gemm(st, w3, ed, 0, T2, L, (ll)ed * L, H3, L, (ll)(4 * ed) * L, 4 * ed, L, ed, 4, 0, 1);
}

extern "C" void kernel_launch(void* const* d_in, const int* in_sizes, int n_in,
                              void* d_out, int out_size, void* d_ws, size_t ws_size,
                              hipStream_t stream) {
  (void)in_sizes; (void)out_size;
  if (n_in < 103) return;
#define IN(i) ((const float*)d_in[(i)])

  const size_t BUFE = 8388608;  // 32 MB fp32 arena
  const size_t need = (6 * BUFE + (1u << 20) + 4096) * sizeof(float);
  if (ws_size < need) return;
  float* W = (float*)d_ws;
  float* B0 = W + 0 * BUFE;
  float* B1 = W + 1 * BUFE;
  float* B2 = W + 2 * BUFE;
  float* B3 = W + 3 * BUFE;
  float* B4 = W + 4 * BUFE;
  float* B5 = W + 5 * BUFE;
  float* WPK = W + 6 * BUFE;       // repacked k3 weights
  float* SC = WPK + (1u << 20);    // BN scale
  float* SH = SC + 2048;           // BN shift

  // ---------------- Stem ----------------
  {
    const int L = 16384;
    stem_conv_kernel<<<CEILDIV(4 * 64 * L, 256), 256, 0, stream>>>(
        IN(0), IN(1), IN(2), IN(3), IN(4), B0, L);
    ln_ch_kernel<<<4 * L, 128, 0, stream>>>(B0, IN(5), IN(6), 64, L);
    relu_copy_kernel<<<CEILDIV(4 * 64 * L, 256), 256, 0, stream>>>(B0, B1, 4 * 64 * L);
    // out = relu(convW @ relu(h) + h)
    gemm(stream, IN(7), 64, 0, B1, L, (ll)64 * L, B2, L, (ll)64 * L,
         64, L, 64, 4, EPI_RESID | EPI_RELU, 1, nullptr, B0, (ll)64 * L);
    // attention pool: logits = pool_w @ out
    gemm(stream, IN(8), 64, 0, B2, L, (ll)64 * L, B3, L, (ll)64 * L, 64, L, 64, 4, 0, 1);
    pool_combine_kernel<<<CEILDIV(4 * 64 * (L / 2), 256), 256, 0, stream>>>(B2, B3, B0, 64, L);
  }

  // ---------------- ResNet stages ----------------
  const int Ls[3] = {8192, 4096, 2048};
  const int cins[3] = {64, 256, 512};
  for (int s = 0; s < 3; ++s) {
    const int base = 9 + 18 * s;
    const int ed = 64 << s, co4 = 4 * ed, cin = cins[s], L = Ls[s];
    const int nco4 = 4 * co4 * L;  // elements of a (4, co4, L) tensor
    // block1
    run_bneck(stream, B0, cin, IN(base + 0), IN(base + 1), IN(base + 2),
              IN(base + 3), IN(base + 4), IN(base + 5), IN(base + 6),
              ed, L, B1, B2, B3, B5, WPK, SC, SH);
    ln_ch_kernel<<<4 * L, 128, 0, stream>>>(B3, IN(base + 8), IN(base + 9), co4, L);
    // shortcut
    gemm(stream, IN(base + 7), cin, 0, B0, L, (ll)cin * L, B4, L, (ll)co4 * L,
         co4, L, cin, 4, 0, 1);
    ln_ch_kernel<<<4 * L, 128, 0, stream>>>(B4, IN(base + 8), IN(base + 9), co4, L);
    add_relu_kernel<<<CEILDIV(nco4, 256), 256, 0, stream>>>(B3, B4, nco4);  // out -> B4
    // block2 (one residual bottleneck)
    const int b2i = base + 10;
    run_bneck(stream, B4, co4, IN(b2i + 0), IN(b2i + 1), IN(b2i + 2),
              IN(b2i + 3), IN(b2i + 4), IN(b2i + 5), IN(b2i + 6),
              ed, L, B1, B2, B3, B5, WPK, SC, SH);
    ln_ch_kernel<<<4 * L, 128, 0, stream>>>(B3, IN(base + 8), IN(base + 9), co4, L);
    add_relu_kernel<<<CEILDIV(nco4, 256), 256, 0, stream>>>(B3, B4, nco4);  // out -> B4
    // pool
    gemm(stream, IN(base + 17), co4, 0, B4, L, (ll)co4 * L, B5, L, (ll)co4 * L,
         co4, L, co4, 4, 0, 1);
    pool_combine_kernel<<<CEILDIV(4 * co4 * (L / 2), 256), 256, 0, stream>>>(B4, B5, B0, co4, L);
  }

  // ---------------- crop + transpose -> (4, 896, 1024) ----------------
  crop_transpose_kernel<<<CEILDIV(4 * 896 * 1024, 256), 256, 0, stream>>>(B0, B1);
  float* X = B1;
  const int Sq = 896, D = 1024;
  const ll sX = (ll)Sq * D;

  // ---------------- Transformer ----------------
  for (int layer = 0; layer < 4; ++layer) {
    const float* wq = IN(63 + 4 * layer + 0);
    const float* wk = IN(63 + 4 * layer + 1);
    const float* wv = IN(63 + 4 * layer + 2);
    const float* wo = IN(63 + 4 * layer + 3);
    row_ln_kernel<<<4 * Sq, 256, 0, stream>>>(X, B2, nullptr, nullptr, D, 1);
    gemm(stream, B2, D, sX, wq, D, 0, B3, D, sX, Sq, D, D, 4, 0, 1);
    gemm(stream, B2, D, sX, wk, D, 0, B4, D, sX, Sq, D, D, 4, 0, 1);
    gemm(stream, B2, D, sX, wv, D, 0, B5, D, sX, Sq, D, D, 4, 0, 1);
    rope_kernel<<<CEILDIV(4 * Sq * 512, 256), 256, 0, stream>>>(B3, B4, Sq);
    flash_attn_kernel<<<dim3(Sq / 16, 16, 4), 32, 0, stream>>>(B3, B4, B5, B2, Sq);
    // x = x + attn @ wo
    gemm(stream, B2, D, sX, wo, D, 0, X, D, sX, Sq, D, D, 4, EPI_RESID, 1,
         nullptr, X, sX);

    const float* fg = IN(79 + 6 * layer + 0);
    const float* fb = IN(79 + 6 * layer + 1);
    const float* fw1 = IN(79 + 6 * layer + 2);
    const float* fb1 = IN(79 + 6 * layer + 3);
    const float* fw2 = IN(79 + 6 * layer + 4);
    const float* fb2 = IN(79 + 6 * layer + 5);
    row_ln_kernel<<<4 * Sq, 256, 0, stream>>>(X, B2, fg, fb, D, 0);
    gemm(stream, B2, D, sX, fw1, 2048, 0, B3, 2048, (ll)Sq * 2048,
         Sq, 2048, D, 4, EPI_BIAS | EPI_RELU, 1, fb1);
    float* Cout = (layer == 3) ? (float*)d_out : X;
    gemm(stream, B3, 2048, (ll)Sq * 2048, fw2, D, 0, Cout, D, sX,
         Sq, D, 2048, 4, EPI_BIAS | EPI_RESID, 1, fb2, X, sX);
  }
#undef IN
}
